// QuantScalarMLP_14130442404365
// MI455X (gfx1250) — 16-bit and 8-bit matrix operand order repaired and checked by reading, compiled, not run on hardware
//
#include <hip/hip_runtime.h>

typedef __attribute__((ext_vector_type(16))) _Float16 v16h;
typedef __attribute__((ext_vector_type(8)))  _Float16 v8h;
typedef __attribute__((ext_vector_type(4)))  float    v4f;
typedef __attribute__((ext_vector_type(8)))  float    v8f;
typedef __attribute__((ext_vector_type(2)))  int      v2i;
typedef __attribute__((ext_vector_type(4)))  int      v4i;
typedef __attribute__((ext_vector_type(8)))  int      v8i;

#define D 128
#define ROWS_PER_BLOCK 256

#if defined(__HIP_DEVICE_COMPILE__) && defined(__has_builtin)
#if __has_builtin(__builtin_amdgcn_global_load_async_to_lds_b128)
#define HAVE_ASYNC_LDS 1
typedef __attribute__((address_space(1))) v4i gv4i;
typedef __attribute__((address_space(3))) v4i lv4i;
#endif
#endif

__global__ __launch_bounds__(256) void kabsmax(const float* __restrict__ x,
                                               int n4, unsigned int* out) {
  __shared__ float red[256];
  float m = 0.f;
  const int stride = gridDim.x * blockDim.x;
  for (int i = blockIdx.x * 256 + threadIdx.x; i < n4; i += stride) {
    v4f v = ((const v4f*)x)[i];
    m = fmaxf(m, fmaxf(fmaxf(fabsf(v.x), fabsf(v.y)),
                       fmaxf(fabsf(v.z), fabsf(v.w))));
  }
  red[threadIdx.x] = m;
  __syncthreads();
  for (int s = 128; s > 0; s >>= 1) {
    if (threadIdx.x < s)
      red[threadIdx.x] = fmaxf(red[threadIdx.x], red[threadIdx.x + s]);
    __syncthreads();
  }
  if (threadIdx.x == 0) atomicMax(out, __float_as_uint(red[0]));
}

__global__ __launch_bounds__(256) void kprep(const float* __restrict__ w0,
                                             const float* __restrict__ w1,
                                             const float* __restrict__ w2,
                                             const float* __restrict__ t,
                                             const unsigned int* __restrict__ maxbits,
                                             float* __restrict__ scal,
                                             signed char* __restrict__ w0q,
                                             _Float16* __restrict__ w1h,
                                             _Float16* __restrict__ w2h) {
  __shared__ float red[256];
  const float alpha = 0.08838834764831845f;
  float m = 0.f;
  for (int i = threadIdx.x; i < D * D; i += 256)
    m = fmaxf(m, fabsf(w0[i] * alpha));
  red[threadIdx.x] = m;
  __syncthreads();
  for (int s = 128; s > 0; s >>= 1) {
    if (threadIdx.x < s)
      red[threadIdx.x] = fmaxf(red[threadIdx.x], red[threadIdx.x + s]);
    __syncthreads();
  }
  const float s_w0 = red[0] / 127.f + 1e-12f;
  const float s_x  = __uint_as_float(*maxbits) / 127.f + 1e-12f;
  if (threadIdx.x == 0) {
    scal[0] = 1.f / s_x;
    scal[1] = s_x * t[0] * s_w0;
  }
  const float inv_w0 = 1.f / s_w0;
  for (int i = threadIdx.x; i < D * D; i += 256) {
    const int k = i >> 7, n = i & 127;
    const float q = rintf(fminf(fmaxf(w0[i] * alpha * inv_w0, -127.f), 127.f));
    w0q[n * D + k] = (signed char)q;
    w1h[n * D + k] = (_Float16)(w1[i] * alpha);
    w2h[n * D + k] = (_Float16)(w2[i] * alpha);
  }
}

#define SMEM_W0Q 0
#define SMEM_W1H 16384
#define SMEM_W2H 49152
#define SMEM_SQ  81920
#define SMEM_SA  114688
#define SMEM_BYTES 180224

__global__ __launch_bounds__(256) void kmlp(const float* __restrict__ x,
                                            const signed char* __restrict__ wpack,
                                            const float* __restrict__ scal,
                                            float* __restrict__ out, int nrows) {
  extern __shared__ char smem[];
  signed char* sW0q = (signed char*)(smem + SMEM_W0Q);
  const _Float16* sW1 = (const _Float16*)(smem + SMEM_W1H);
  const _Float16* sW2 = (const _Float16*)(smem + SMEM_W2H);
  signed char* sQ = (signed char*)(smem + SMEM_SQ);
  _Float16* sA = (_Float16*)(smem + SMEM_SA);

  const int tid = threadIdx.x;
  const int wave = tid >> 5;
  const int lane = tid & 31;
  const int lrow = lane & 15;
  const int halfsel = lane >> 4;
  const int rowBase = blockIdx.x * ROWS_PER_BLOCK;

  const float inv_sx = scal[0];
  const float c0 = scal[1];

#ifdef HAVE_ASYNC_LDS
  {
    #pragma unroll
    for (int i = 0; i < 20; ++i) {
      const int idx = (tid + (i << 8)) << 4;
      __builtin_amdgcn_global_load_async_to_lds_b128(
          (gv4i*)(void*)(wpack + idx), (lv4i*)(void*)(smem + idx), 0, 0);
    }
  }
#else
  {
    const v4f* src = (const v4f*)wpack;
    v4f* dst = (v4f*)smem;
    #pragma unroll
    for (int i = 0; i < 20; ++i) dst[tid + (i << 8)] = src[tid + (i << 8)];
  }
#endif

  #pragma unroll
  for (int i = 0; i < 32; ++i) {
    const int idx4 = tid + (i << 8);
    const int row = idx4 >> 5;
    const int c4 = (idx4 & 31) << 2;
    const int grow = rowBase + row;
    v4f v = {0.f, 0.f, 0.f, 0.f};
    if (grow < nrows) v = *(const v4f*)(x + (size_t)grow * D + c4);
    const int q0 = (int)rintf(fminf(fmaxf(v.x * inv_sx, -127.f), 127.f));
    const int q1 = (int)rintf(fminf(fmaxf(v.y * inv_sx, -127.f), 127.f));
    const int q2 = (int)rintf(fminf(fmaxf(v.z * inv_sx, -127.f), 127.f));
    const int q3 = (int)rintf(fminf(fmaxf(v.w * inv_sx, -127.f), 127.f));
    const unsigned packed = ((unsigned)q0 & 255u) | (((unsigned)q1 & 255u) << 8) |
                            (((unsigned)q2 & 255u) << 16) | (((unsigned)q3 & 255u) << 24);
    *(unsigned*)(sQ + (row >> 5) * (32 * D) + (row & 31) * D + c4) = packed;
  }

#ifdef HAVE_ASYNC_LDS
#if __has_builtin(__builtin_amdgcn_s_wait_asynccnt)
  __builtin_amdgcn_s_wait_asynccnt(0);
#else
  asm volatile("s_wait_asynccnt 0" ::: "memory");
#endif
#endif
  __syncthreads();

  const signed char* actq = sQ + wave * (32 * D);
  _Float16* act = sA + wave * (32 * D);

  {
    v8i acc[2][8];
    const v8i zi = {0, 0, 0, 0, 0, 0, 0, 0};
    #pragma unroll
    for (int mt = 0; mt < 2; ++mt)
      #pragma unroll
      for (int n = 0; n < 8; ++n) acc[mt][n] = zi;

    #pragma unroll
    for (int ks = 0; ks < 2; ++ks) {
      v8i a[2];
      #pragma unroll
      for (int mt = 0; mt < 2; ++mt) {
        const signed char* ap = actq + (mt * 16 + lrow) * D + ks * 64 + halfsel * 16;
        const v2i a0 = *(const v2i*)(ap + 0);
        const v2i a1 = *(const v2i*)(ap + 8);
        const v2i a2 = *(const v2i*)(ap + 32);
        const v2i a3 = *(const v2i*)(ap + 40);
        const v4i alo = __builtin_shufflevector(a0, a1, 0, 1, 2, 3);
        const v4i ahi = __builtin_shufflevector(a2, a3, 0, 1, 2, 3);
        a[mt] = __builtin_shufflevector(alo, ahi, 0, 1, 2, 3, 4, 5, 6, 7);
      }
      #pragma unroll
      for (int n = 0; n < 8; ++n) {
        const signed char* bp = sW0q + (n * 16 + lrow) * D + ks * 64 + halfsel * 16;
        const v4i blo = *(const v4i*)(bp + 0);
        const v4i bhi = *(const v4i*)(bp + 32);
        const v8i b = __builtin_shufflevector(blo, bhi, 0, 1, 2, 3, 4, 5, 6, 7);
        acc[0][n] = __builtin_amdgcn_wmma_i32_16x16x64_iu8(true, a[0], true, b,
                                                           acc[0][n], false, false);
        acc[1][n] = __builtin_amdgcn_wmma_i32_16x16x64_iu8(true, a[1], true, b,
                                                           acc[1][n], false, false);
      }
    }
    #pragma unroll
    for (int mt = 0; mt < 2; ++mt)
      #pragma unroll
      for (int n = 0; n < 8; ++n)
        #pragma unroll
        for (int r = 0; r < 8; ++r) {
          float v = (float)acc[mt][n][r] * c0;
          v = v / (1.f + __expf(-v));
          act[(mt * 16 + r + halfsel * 8) * D + n * 16 + lrow] = (_Float16)v;
        }
  }

  #pragma unroll
  for (int layer = 1; layer < 3; ++layer) {
    const _Float16* wl = (layer == 1) ? sW1 : sW2;
    v8f acc[2][8];
    const v8f zf = {0.f, 0.f, 0.f, 0.f, 0.f, 0.f, 0.f, 0.f};
    #pragma unroll
    for (int mt = 0; mt < 2; ++mt)
      #pragma unroll
      for (int n = 0; n < 8; ++n) acc[mt][n] = zf;

    #pragma unroll
    for (int ks = 0; ks < 4; ++ks) {
      v16h a[2];
      #pragma unroll
      for (int mt = 0; mt < 2; ++mt) {
        const _Float16* ap = act + (mt * 16 + lrow) * D + ks * 32 + halfsel * 8;
        const v8h a0 = *(const v8h*)(ap + 0);
        const v8h a1 = *(const v8h*)(ap + 16);
        a[mt] = __builtin_shufflevector(a0, a1, 0, 1, 2, 3, 4, 5, 6, 7, 8, 9, 10,
                                        11, 12, 13, 14, 15);
      }
      #pragma unroll
      for (int n = 0; n < 8; ++n) {
        const _Float16* bp = wl + (n * 16 + lrow) * D + ks * 32 + halfsel * 8;
        const v8h b0 = *(const v8h*)(bp + 0);
        const v8h b1 = *(const v8h*)(bp + 16);
        const v16h b = __builtin_shufflevector(b0, b1, 0, 1, 2, 3, 4, 5, 6, 7, 8,
                                               9, 10, 11, 12, 13, 14, 15);
        acc[0][n] = __builtin_amdgcn_wmma_f32_16x16x32_f16(
            false, a[0], false, b, (short)0, acc[0][n], false, false);
        acc[1][n] = __builtin_amdgcn_wmma_f32_16x16x32_f16(
            false, a[1], false, b, (short)0, acc[1][n], false, false);
      }
    }

    if (layer == 1) {
      #pragma unroll
      for (int mt = 0; mt < 2; ++mt)
        #pragma unroll
        for (int n = 0; n < 8; ++n)
          #pragma unroll
          for (int r = 0; r < 8; ++r) {
            float v = acc[mt][n][r];
            v = v / (1.f + __expf(-v));
            act[(mt * 16 + r + halfsel * 8) * D + n * 16 + lrow] = (_Float16)v;
          }
    } else {
      #pragma unroll
      for (int mt = 0; mt < 2; ++mt) {
        const int growBase = rowBase + wave * 32 + mt * 16 + halfsel * 8;
        #pragma unroll
        for (int n = 0; n < 8; ++n)
          #pragma unroll
          for (int r = 0; r < 8; ++r) {
            const int grow = growBase + r;
            if (grow < nrows)
              out[(size_t)grow * D + n * 16 + lrow] = acc[mt][n][r];
          }
      }
    }
  }
}

extern "C" void kernel_launch(void* const* d_in, const int* in_sizes, int n_in,
                              void* d_out, int out_size, void* d_ws,
                              size_t ws_size, hipStream_t stream) {
  const float* x  = (const float*)d_in[0];
  const float* w0 = (const float*)d_in[1];
  const float* w1 = (const float*)d_in[2];
  const float* w2 = (const float*)d_in[3];
  const float* t  = (const float*)d_in[4];
  float* out = (float*)d_out;

  const int total = in_sizes[0];
  const int nrows = total / D;

  unsigned int* maxbits = (unsigned int*)d_ws;
  float* scal = (float*)((char*)d_ws + 8);
  signed char* wpack = (signed char*)((char*)d_ws + 256);
  signed char* w0q = wpack;
  _Float16* w1h = (_Float16*)(wpack + 16384);
  _Float16* w2h = (_Float16*)(wpack + 49152);

  (void)hipMemsetAsync(d_ws, 0, 16, stream);
  kabsmax<<<2048, 256, 0, stream>>>(x, total / 4, maxbits);
  kprep<<<1, 256, 0, stream>>>(w0, w1, w2, t, maxbits, scal, w0q, w1h, w2h);

  const int nblk = (nrows + ROWS_PER_BLOCK - 1) / ROWS_PER_BLOCK;
  kmlp<<<nblk, 256, SMEM_BYTES, stream>>>(x, wpack, scal, out, nrows);
}
